// HeterogeneousGNN_72584947302719
// MI455X (gfx1250) — compile-verified, matrix operand order checked by reading, not run on hardware
//
#include <hip/hip_runtime.h>
#include <hip/hip_bf16.h>
#include <math.h>

typedef __attribute__((ext_vector_type(16))) _Float16 v16h;
typedef __attribute__((ext_vector_type(8)))  float    v8f;

// ---------------------------------------------------------------- utilities
__device__ __forceinline__ void atomicMaxF(float* a, float v) {
  if (v >= 0.f) atomicMax((int*)a, __float_as_int(v));
  else          atomicMin((unsigned int*)a, __float_as_uint(v));
}

__global__ void fill_kernel(float* p, float v, long n) {
  long i = (long)blockIdx.x * blockDim.x + threadIdx.x;
  long stride = (long)gridDim.x * blockDim.x;
  for (; i < n; i += stride) p[i] = v;
}

// ------------------------------------------------ column mean / 1/(std+eps)
__global__ void colstats_kernel(const float* __restrict__ x, int n, int d,
                                float* __restrict__ mean, float* __restrict__ rstd) {
  int c = blockIdx.x;
  if (c >= d) return;
  float s = 0.f, sq = 0.f;
  for (int r = threadIdx.x; r < n; r += blockDim.x) {
    float v = x[(long)r * d + c];
    s += v; sq += v * v;
  }
  __shared__ float rs[256], rq[256];
  rs[threadIdx.x] = s; rq[threadIdx.x] = sq;
  __syncthreads();
  for (int st = 128; st > 0; st >>= 1) {
    if ((int)threadIdx.x < st) { rs[threadIdx.x] += rs[threadIdx.x + st];
                                 rq[threadIdx.x] += rq[threadIdx.x + st]; }
    __syncthreads();
  }
  if (threadIdx.x == 0) {
    float m = rs[0] / n;
    float var = (rq[0] - n * m * m) / (n - 1);   // ddof=1
    var = var > 0.f ? var : 0.f;
    mean[c] = m;
    rstd[c] = 1.f / (sqrtf(var) + 1e-6f);
  }
}

// ------------------------------------- fused standardize + linear embedding
__global__ void embed_kernel(const float* __restrict__ x, int n, int d,
                             const float* __restrict__ mean, const float* __restrict__ rstd,
                             const float* __restrict__ sg, const float* __restrict__ sb,
                             const float* __restrict__ W, const float* __restrict__ bias,
                             float* __restrict__ h) {
  long t = (long)blockIdx.x * blockDim.x + threadIdx.x;
  if (t >= (long)n * 128) return;
  int row = (int)(t >> 7), c = (int)(t & 127);
  float s = bias[c];
  for (int j = 0; j < d; ++j) {
    float xs = (x[(long)row * d + j] - mean[j]) * rstd[j] * sg[j] + sb[j];
    s += xs * W[j * 128 + c];
  }
  h[t] = s;
}

// ---------------------------------------------------------- WMMA f16 GEMM
// hs[n x 128] = A[n x 128] @ W[128 x 128]     (fp32 in, fp16 WMMA, fp32 acc)
__global__ __launch_bounds__(256)
void gemm128_wmma_kernel(const float* __restrict__ A, const float* __restrict__ W,
                         float* __restrict__ C, int n) {
  __shared__ _Float16 Ah[64 * 128];     // 16 KB  (64 rows of A)
  __shared__ _Float16 WT[128 * 128];    // 32 KB  (W transposed: WT[n][k])
  const int tid  = threadIdx.x;
  const int row0 = blockIdx.x * 64;

  for (int i = tid; i < 128 * 128; i += 256) {
    int k = i >> 7, nn = i & 127;
    WT[nn * 128 + k] = (_Float16)W[i];
  }
  for (int i = tid; i < 64 * 128; i += 256) {
    int r = i >> 7, k = i & 127;
    int gr = row0 + r;
    Ah[i] = (gr < n) ? (_Float16)A[(long)gr * 128 + k] : (_Float16)0.f;
  }
  __syncthreads();

  const int wave  = tid >> 5;
  const int lane  = tid & 31;
  const int mtile = wave >> 1;      // 0..3
  const int nhalf = wave & 1;       // cols [nhalf*64, nhalf*64+64)
  const int lm = lane & 15, lh = lane >> 4;
  const int arow = mtile * 16 + lm;

  v8f acc[4] = {};
#pragma unroll
  for (int kt = 0; kt < 4; ++kt) {        // fully unrolled: distinct regs per
    const int k0 = kt * 32;               // step -> scheduler hides WMMA hazards
    v16h a;
    const _Float16* ap = &Ah[arow * 128 + k0];
#pragma unroll
    for (int v = 0; v < 8; ++v) {
      int k = (v < 4) ? (8 * lh + 2 * v) : (16 + 8 * lh + 2 * (v - 4));
      a[2 * v]     = ap[k];
      a[2 * v + 1] = ap[k + 1];
    }
#pragma unroll
    for (int nt = 0; nt < 4; ++nt) {
      const int ncol = nhalf * 64 + nt * 16 + lm;
      const _Float16* bp = &WT[ncol * 128 + k0];
      v16h b;
#pragma unroll
      for (int v = 0; v < 8; ++v) {
        int k = (v < 4) ? (8 * lh + 2 * v) : (16 + 8 * lh + 2 * (v - 4));
        b[2 * v]     = bp[k];
        b[2 * v + 1] = bp[k + 1];
      }
      acc[nt] = __builtin_amdgcn_wmma_f32_16x16x32_f16(
          false, a, false, b, (short)0, acc[nt], false, false);
    }
  }

  // epilogue: branch-free fast path for full tiles (vast majority of blocks)
  if (row0 + 64 <= n) {
#pragma unroll
    for (int nt = 0; nt < 4; ++nt) {
#pragma unroll
      for (int i = 0; i < 8; ++i) {
        int r = row0 + mtile * 16 + lh * 8 + i;
        int c = nhalf * 64 + nt * 16 + lm;
        C[(long)r * 128 + c] = acc[nt][i];
      }
    }
  } else {
#pragma unroll
    for (int nt = 0; nt < 4; ++nt) {
#pragma unroll
      for (int i = 0; i < 8; ++i) {
        int r = row0 + mtile * 16 + lh * 8 + i;
        int c = nhalf * 64 + nt * 16 + lm;
        if (r < n) C[(long)r * 128 + c] = acc[nt][i];
      }
    }
  }
}

// ------------------------------- fold Wdst x a_dst -> wda[128x4], Wedge x a_edge
__global__ void fold_kernel(const float* __restrict__ Wdst, const float* __restrict__ a_dst,
                            const float* __restrict__ Wedge, const float* __restrict__ a_edge,
                            int ed, float* __restrict__ wda, float* __restrict__ ced) {
  int t = blockIdx.x * blockDim.x + threadIdx.x;
  if (t < 512) {
    int k = t >> 2, hh = t & 3;
    float s = 0.f;
    for (int dd = 0; dd < 32; ++dd) s += Wdst[k * 128 + hh * 32 + dd] * a_dst[hh * 32 + dd];
    wda[t] = s;
  } else if (t < 512 + ed * 4) {
    int u = t - 512; int j = u >> 2, hh = u & 3;
    float s = 0.f;
    for (int dd = 0; dd < 32; ++dd) s += Wedge[j * 128 + hh * 32 + dd] * a_edge[hh * 32 + dd];
    ced[u] = s;
  }
}

// d[n,h] = sum_k hs[n, h*32+k] * a_src[h,k]
__global__ void headdot_kernel(const float* __restrict__ hs, const float* __restrict__ a,
                               float* __restrict__ out, int n) {
  long t = (long)blockIdx.x * blockDim.x + threadIdx.x;
  if (t >= (long)n * 4) return;
  int node = (int)(t >> 2), hh = (int)(t & 3);
  const float* p = hs + (long)node * 128 + hh * 32;
  const float* q = a + hh * 32;
  float s = 0.f;
  for (int k = 0; k < 32; ++k) s += p[k] * q[k];
  out[t] = s;
}

// d[n,h] = sum_k h[n,k] * wda[k,h]
__global__ void folddot_kernel(const float* __restrict__ h, const float* __restrict__ wda,
                               float* __restrict__ out, int n) {
  long t = (long)blockIdx.x * blockDim.x + threadIdx.x;
  if (t >= (long)n * 4) return;
  int node = (int)(t >> 2), hh = (int)(t & 3);
  const float* p = h + (long)node * 128;
  float s = 0.f;
  for (int k = 0; k < 128; ++k) s += p[k] * wda[k * 4 + hh];
  out[t] = s;
}

// ------------------------------------------- logits + segment max (atomic)
__global__ void logits_max_kernel(const int* __restrict__ ei, int E,
                                  const float* __restrict__ ef, int ed,
                                  const float* __restrict__ em, const float* __restrict__ er,
                                  const float* __restrict__ eg, const float* __restrict__ eb,
                                  const float* __restrict__ ced,
                                  const float* __restrict__ dsrc, const float* __restrict__ ddst,
                                  float* __restrict__ elog, float* __restrict__ mmax) {
  long t = (long)blockIdx.x * blockDim.x + threadIdx.x;
  if (t >= (long)E * 4) return;
  int e = (int)(t >> 2), hh = (int)(t & 3);
  int src = ei[e], dst = ei[E + e];
  float s = dsrc[src * 4 + hh] + ddst[dst * 4 + hh];
  for (int j = 0; j < ed; ++j) {
    float xs = (ef[(long)e * ed + j] - em[j]) * er[j] * eg[j] + eb[j];
    s += xs * ced[j * 4 + hh];
  }
  s = (s > 0.f) ? s : 0.2f * s;   // leaky_relu 0.2
  elog[t] = s;
  atomicMaxF(&mmax[dst * 4 + hh], s);
}

// ---------------------------------------------- exp(e - max) + segment sum
__global__ void expsum_kernel(const int* __restrict__ ei, int E,
                              float* __restrict__ elog,
                              const float* __restrict__ mmax, float* __restrict__ ssum) {
  long t = (long)blockIdx.x * blockDim.x + threadIdx.x;
  if (t >= (long)E * 4) return;
  int e = (int)(t >> 2), hh = (int)(t & 3);
  int dst = ei[E + e];
  float v = expf(elog[t] - mmax[dst * 4 + hh]);
  elog[t] = v;
  atomicAdd(&ssum[dst * 4 + hh], v);
}

// -------------------------------------- wave-per-edge message scatter (f32 atomics)
__global__ __launch_bounds__(256)
void scatter_kernel(const int* __restrict__ ei, int E,
                    const float* __restrict__ elog, const float* __restrict__ ssum,
                    const float* __restrict__ hs, float* __restrict__ agg) {
  int gw   = (int)(((long)blockIdx.x * blockDim.x + threadIdx.x) >> 5);
  int lane = threadIdx.x & 31;
  int nw   = (int)(((long)gridDim.x * blockDim.x) >> 5);
  for (int e = gw; e < E; e += nw) {
    // prefetch next edge's source row down the grid-stride chain
    // (gather is the HBM-bound hot loop; emits global_prefetch_b8)
    int en = e + nw;
    if (en < E) {
      __builtin_prefetch(&hs[(long)ei[en] * 128 + lane], 0, 1);
      __builtin_prefetch(&elog[(long)en * 4], 0, 1);
    }
    int src = ei[e], dst = ei[E + e];
    const float* hp = hs + (long)src * 128;
    float* ap = agg + (long)dst * 128;
#pragma unroll
    for (int hh = 0; hh < 4; ++hh) {
      float alpha = elog[(long)e * 4 + hh] / (ssum[dst * 4 + hh] + 1e-8f);
      int c = hh * 32 + lane;
      atomicAdd(&ap[c], alpha * hp[c]);
    }
  }
}

// ----------------------------------------- residual + LayerNorm (in place)
__global__ __launch_bounds__(128)
void resid_ln_kernel(float* __restrict__ h, const float* __restrict__ agg,
                     const float* __restrict__ g, const float* __restrict__ b, int n) {
  int row = blockIdx.x;
  if (row >= n) return;
  int c = threadIdx.x;
  __shared__ float red[128];
  __shared__ float sm, sv;
  float x = h[(long)row * 128 + c] + agg[(long)row * 128 + c];
  red[c] = x; __syncthreads();
  for (int s = 64; s > 0; s >>= 1) { if (c < s) red[c] += red[c + s]; __syncthreads(); }
  if (c == 0) sm = red[0] * (1.f / 128.f);
  __syncthreads();
  float d = x - sm;
  red[c] = d * d; __syncthreads();
  for (int s = 64; s > 0; s >>= 1) { if (c < s) red[c] += red[c + s]; __syncthreads(); }
  if (c == 0) sv = red[0] * (1.f / 128.f);
  __syncthreads();
  h[(long)row * 128 + c] = d * rsqrtf(sv + 1e-5f) * g[c] + b[c];
}

// -------------------------------------------------- column mean (pooling)
__global__ void colmean_kernel(const float* __restrict__ h, int n, float* __restrict__ out) {
  int c = blockIdx.x;   // 0..127
  float s = 0.f;
  for (int r = threadIdx.x; r < n; r += blockDim.x) s += h[(long)r * 128 + c];
  __shared__ float red[256];
  red[threadIdx.x] = s; __syncthreads();
  for (int st = 128; st > 0; st >>= 1) {
    if ((int)threadIdx.x < st) red[threadIdx.x] += red[threadIdx.x + st];
    __syncthreads();
  }
  if (threadIdx.x == 0) out[c] = red[0] / n;
}

// ----------------------------------------------------------------- MLP head
__global__ __launch_bounds__(128)
void mlp_kernel(const float* __restrict__ gf,
                const float* __restrict__ W1, const float* __restrict__ b1,
                const float* __restrict__ W2, const float* __restrict__ b2,
                float* __restrict__ gs) {
  __shared__ float hid[128];
  int c = threadIdx.x;
  float s = b1[c];
  for (int j = 0; j < 512; ++j) s += gf[j] * W1[j * 128 + c];
  hid[c] = s > 0.f ? s : 0.f;
  __syncthreads();
  float o = b2[c];
  for (int j = 0; j < 128; ++j) o += hid[j] * W2[j * 128 + c];
  gs[c] = o;
}

// ======================================================================
extern "C" void kernel_launch(void* const* d_in, const int* in_sizes, int n_in,
                              void* d_out, int out_size, void* d_ws, size_t ws_size,
                              hipStream_t stream) {
  (void)n_in; (void)out_size; (void)ws_size;
#define P(i) ((const float*)d_in[(i)])
#define PI(i) ((const int*)d_in[(i)])
  const int NO = in_sizes[0] / 16, NA = in_sizes[1] / 12;
  const int NW = in_sizes[2] / 8,  NB = in_sizes[3] / 8;
  const int EOA = in_sizes[7] / 2, EAW = in_sizes[9] / 2, EAB = in_sizes[11] / 2;

  float* out = (float*)d_out;
  float* h_o = out;
  float* h_a = h_o + (long)NO * 128;
  float* h_w = h_a + (long)NA * 128;
  float* h_b = h_w + (long)NW * 128;
  float* gs  = h_b + (long)NB * 128;

  // ---- workspace carve
  float* ws = (float*)d_ws;
  float* agg_o = ws; ws += (long)NO * 128;
  float* agg_a = ws; ws += (long)NA * 128;
  float* agg_w = ws; ws += (long)NW * 128;
  float* agg_b = ws; ws += (long)NB * 128;
  float* hs    = ws; ws += (long)NO * 128;   // max src-node count
  float* dsrc  = ws; ws += (long)NO * 4;
  float* ddst  = ws; ws += (long)NO * 4;     // max dst-node count (a2o)
  float* elog  = ws; ws += (long)EOA * 4;    // max edge count
  float* mmax  = ws; ws += (long)NO * 4;
  float* ssum  = ws; ws += (long)NO * 4;
  float* stats = ws; ws += 128;              // mean/rstd pairs
  float* wda   = ws; ws += 512;
  float* ced   = ws; ws += 16;
  float* gf    = ws; ws += 512;

  float* m_ord = stats + 0,  * r_ord = stats + 16;
  float* m_agv = stats + 32, * r_agv = stats + 44;
  float* m_wai = stats + 56, * r_wai = stats + 64;
  float* m_buf = stats + 72, * r_buf = stats + 80;
  float* m_eoa = stats + 88, * r_eoa = stats + 89;
  float* m_eaw = stats + 90, * r_eaw = stats + 92;
  float* m_eab = stats + 94, * r_eab = stats + 95;

  // ---- 1. column stats (7 feature sets)
  colstats_kernel<<<16, 256, 0, stream>>>(P(0), NO, 16, m_ord, r_ord);
  colstats_kernel<<<12, 256, 0, stream>>>(P(1), NA, 12, m_agv, r_agv);
  colstats_kernel<<<8,  256, 0, stream>>>(P(2), NW, 8,  m_wai, r_wai);
  colstats_kernel<<<8,  256, 0, stream>>>(P(3), NB, 8,  m_buf, r_buf);
  colstats_kernel<<<1,  256, 0, stream>>>(P(4), EOA, 1, m_eoa, r_eoa);
  colstats_kernel<<<2,  256, 0, stream>>>(P(5), EAW, 2, m_eaw, r_eaw);
  colstats_kernel<<<1,  256, 0, stream>>>(P(6), EAB, 1, m_eab, r_eab);

  // ---- 2. fused standardize + embed (params: embed.{agv,buffer,order,waiting})
  embed_kernel<<<(int)(((long)NO * 128 + 255) / 256), 256, 0, stream>>>(
      P(0), NO, 16, m_ord, r_ord, P(124), P(123), P(17), P(18), h_o);
  embed_kernel<<<(int)(((long)NA * 128 + 255) / 256), 256, 0, stream>>>(
      P(1), NA, 12, m_agv, r_agv, P(114), P(113), P(13), P(14), h_a);
  embed_kernel<<<(int)(((long)NW * 128 + 255) / 256), 256, 0, stream>>>(
      P(2), NW, 8, m_wai, r_wai, P(126), P(125), P(19), P(20), h_w);
  embed_kernel<<<(int)(((long)NB * 128 + 255) / 256), 256, 0, stream>>>(
      P(3), NB, 8, m_buf, r_buf, P(116), P(115), P(15), P(16), h_b);

  // ---- relation descriptors (gat pytree sorted: a2b,a2o,a2w,b2a,o2a,w2a)
  struct Rel {
    const float* hsrc; int ns; const float* hdst; int nd; float* agg;
    const int* ei; int E; const float* ef; int ed;
    const float* em; const float* er; const float* eg; const float* eb;
    int pb[2];
  };
  Rel rels[6] = {
    { h_o, NO, h_a, NA, agg_a, PI(7),  EOA, P(4), 1, m_eoa, r_eoa, P(122), P(121), {69, 75} }, // o2a
    { h_a, NA, h_o, NO, agg_o, PI(8),  EOA, P(4), 1, m_eoa, r_eoa, P(122), P(121), {33, 39} }, // a2o
    { h_a, NA, h_w, NW, agg_w, PI(9),  EAW, P(5), 2, m_eaw, r_eaw, P(120), P(119), {45, 51} }, // a2w
    { h_w, NW, h_a, NA, agg_a, PI(10), EAW, P(5), 2, m_eaw, r_eaw, P(120), P(119), {81, 87} }, // w2a
    { h_a, NA, h_b, NB, agg_b, PI(11), EAB, P(6), 1, m_eab, r_eab, P(118), P(117), {21, 27} }, // a2b
    { h_b, NB, h_a, NA, agg_a, PI(12), EAB, P(6), 1, m_eab, r_eab, P(118), P(117), {57, 63} }, // b2a
  };

  for (int layer = 0; layer < 2; ++layer) {
    // zero aggregation buffers
    fill_kernel<<<2048, 256, 0, stream>>>(agg_o, 0.f, (long)NO * 128);
    fill_kernel<<<512,  256, 0, stream>>>(agg_a, 0.f, (long)NA * 128);
    fill_kernel<<<512,  256, 0, stream>>>(agg_w, 0.f, (long)NW * 128);
    fill_kernel<<<512,  256, 0, stream>>>(agg_b, 0.f, (long)NB * 128);

    for (int r = 0; r < 6; ++r) {
      const Rel& R = rels[r];
      int pb = R.pb[layer];
      // gat_p sorted: +0 Wdst, +1 Wedge, +2 Wsrc, +3 a_dst, +4 a_edge, +5 a_src
      const float* Wdst = P(pb + 0); const float* Wedge = P(pb + 1);
      const float* Wsrc = P(pb + 2); const float* a_dst = P(pb + 3);
      const float* a_edge = P(pb + 4); const float* a_src = P(pb + 5);

      fold_kernel<<<3, 256, 0, stream>>>(Wdst, a_dst, Wedge, a_edge, R.ed, wda, ced);
      // hs = h_src @ Wsrc   (WMMA)
      gemm128_wmma_kernel<<<(R.ns + 63) / 64, 256, 0, stream>>>(R.hsrc, Wsrc, hs, R.ns);
      headdot_kernel<<<(int)(((long)R.ns * 4 + 255) / 256), 256, 0, stream>>>(hs, a_src, dsrc, R.ns);
      folddot_kernel<<<(int)(((long)R.nd * 4 + 255) / 256), 256, 0, stream>>>(R.hdst, wda, ddst, R.nd);
      fill_kernel<<<256, 256, 0, stream>>>(mmax, -1e30f, (long)R.nd * 4);
      fill_kernel<<<256, 256, 0, stream>>>(ssum, 0.f,    (long)R.nd * 4);
      long e4 = (long)R.E * 4;
      logits_max_kernel<<<(int)((e4 + 255) / 256), 256, 0, stream>>>(
          R.ei, R.E, R.ef, R.ed, R.em, R.er, R.eg, R.eb, ced, dsrc, ddst, elog, mmax);
      expsum_kernel<<<(int)((e4 + 255) / 256), 256, 0, stream>>>(R.ei, R.E, elog, mmax, ssum);
      scatter_kernel<<<(R.E + 7) / 8, 256, 0, stream>>>(R.ei, R.E, elog, ssum, hs, R.agg);
    }

    // residual + LN  (ln sorted: agv@93, buffer@97, order@101, waiting@105; {b,g} per layer)
    resid_ln_kernel<<<NO, 128, 0, stream>>>(h_o, agg_o, P(101 + 2 * layer + 1), P(101 + 2 * layer), NO);
    resid_ln_kernel<<<NA, 128, 0, stream>>>(h_a, agg_a, P(93  + 2 * layer + 1), P(93  + 2 * layer), NA);
    resid_ln_kernel<<<NW, 128, 0, stream>>>(h_w, agg_w, P(105 + 2 * layer + 1), P(105 + 2 * layer), NW);
    resid_ln_kernel<<<NB, 128, 0, stream>>>(h_b, agg_b, P(97  + 2 * layer + 1), P(97  + 2 * layer), NB);
  }

  // ---- pooling + MLP  (mlp sorted: W1@109, W2@110, b1@111, b2@112)
  colmean_kernel<<<128, 256, 0, stream>>>(h_o, NO, gf + 0);
  colmean_kernel<<<128, 256, 0, stream>>>(h_a, NA, gf + 128);
  colmean_kernel<<<128, 256, 0, stream>>>(h_w, NW, gf + 256);
  colmean_kernel<<<128, 256, 0, stream>>>(h_b, NB, gf + 384);
  mlp_kernel<<<1, 128, 0, stream>>>(gf, P(109), P(111), P(110), P(112), gs);
#undef P
#undef PI
}
